// LlamaAttention_16329465659655
// MI455X (gfx1250) — matrix operand order repaired and checked by reading, not compiled, not run on hardware
//
#include <hip/hip_runtime.h>

// ---------------------------------------------------------------------------
// MI455X (gfx1250) Llama attention layer.
// fp32 -> bf16 convert, then all GEMMs + attention via v_wmma_f32_16x16x32_bf16.
// GEMM B-panels staged into LDS by the Tensor Data Mover (tensor_load_to_lds,
// s_wait_tensorcnt), double-buffered and shared by the 8 waves of a block.
// S=2048, H=32, KVH=8, D=128, HID=4096.
// ---------------------------------------------------------------------------

typedef __attribute__((ext_vector_type(16))) __bf16         v16bf;
typedef __attribute__((ext_vector_type(8)))  float          v8f;
typedef __attribute__((ext_vector_type(4)))  unsigned int   u32x4;
typedef __attribute__((ext_vector_type(4)))  float          f32x4;
typedef __attribute__((ext_vector_type(4)))  unsigned short u16x4;
typedef __attribute__((ext_vector_type(8)))  int            i32x8;
typedef __attribute__((ext_vector_type(4)))  int            i32x4;

#define S_LEN   2048
#define NHEAD   32
#define NKVH    8
#define HDIM    128
#define HID     4096
#define KVHID   1024

#if defined(__has_builtin)
#  if __has_builtin(__builtin_amdgcn_tensor_load_to_lds) && \
      __has_builtin(__builtin_amdgcn_s_wait_tensorcnt)
#    define USE_TDM 1
#  endif
#endif
#ifndef USE_TDM
#  define USE_TDM 0
#endif

union FragAB { u32x4 q[2]; v16bf v; };

__device__ __forceinline__ unsigned short f2bf(float f) {
    union { float f; unsigned u; } x; x.f = f;
    unsigned r = x.u + 0x7FFFu + ((x.u >> 16) & 1u);   // round-to-nearest-even
    return (unsigned short)(r >> 16);
}
__device__ __forceinline__ float bf2f(unsigned short b) {
    union { unsigned u; float f; } x; x.u = ((unsigned)b) << 16;
    return x.f;
}

// ds_swizzle xor-reduction step: stays within each 16-lane group for mask<16.
#define SWZ_MAX(v, m) { union { float f; int i; } _x; _x.f = (v);            \
    _x.i = __builtin_amdgcn_ds_swizzle(_x.i, 0x1F | ((m) << 10));            \
    (v) = fmaxf((v), _x.f); }
#define SWZ_ADD(v, m) { union { float f; int i; } _x; _x.f = (v);            \
    _x.i = __builtin_amdgcn_ds_swizzle(_x.i, 0x1F | ((m) << 10));            \
    (v) += _x.f; }

// ---------------------------------------------------------------------------
// fp32 -> bf16 conversion (4 elements / thread, b128 loads)
// ---------------------------------------------------------------------------
__global__ __launch_bounds__(256) void cvt_bf16(const float* __restrict__ in,
                                                unsigned short* __restrict__ out,
                                                int n) {
    int i = (blockIdx.x * 256 + threadIdx.x) * 4;
    if (i + 3 < n) {
        f32x4 v = *(const f32x4*)(in + i);
        u16x4 o;
        o.x = f2bf(v.x); o.y = f2bf(v.y); o.z = f2bf(v.z); o.w = f2bf(v.w);
        *(u16x4*)(out + i) = o;
    }
}

#if USE_TDM
// ---------------------------------------------------------------------------
// TDM: stage a [tile_n=64 rows x tile_k=32 cols] bf16 panel of W into LDS.
// D# per CDNA5 ISA ch.8: group0 = {count=1 | lds_addr | global_addr | type=2},
// group1 = {data_size=1(2B), tensor_dim0/1, tile_dim0/1, dim0_stride}.
// All fields block-uniform -> SGPRs.
// ---------------------------------------------------------------------------
__device__ __forceinline__ void tdm_load_panel(unsigned lds_off,
                                               const unsigned short* gsrc,
                                               int k_total, int stride_elems) {
    unsigned long long ga = (unsigned long long)gsrc;
    u32x4 g0;
    g0.x = 1u;                                        // count=1, user mode
    g0.y = lds_off;                                   // lds_addr (bytes)
    g0.z = (unsigned)(ga & 0xFFFFFFFFu);              // global_addr[31:0]
    g0.w = (unsigned)((ga >> 32) & 0x01FFFFFFu)       // global_addr[56:32]
         | (2u << 30);                                // type = 2 ("image")
    i32x8 g1;
    g1[0] = 1 << 16;                                  // data_size = 1 -> 2 bytes
    g1[1] = (k_total & 0xFFFF) << 16;                 // tensor_dim0[15:0]
    g1[2] = ((64 & 0xFFFF) << 16)                     // tensor_dim1[15:0] = 64 rows
          | ((k_total >> 16) & 0xFFFF);               // tensor_dim0[31:16]
    g1[3] = (32 << 16);                               // tile_dim0 = 32 (+dim1 hi = 0)
    g1[4] = 64;                                       // tile_dim1 = 64, tile_dim2 = 0
    g1[5] = stride_elems;                             // tensor_dim0_stride[31:0]
    g1[6] = 0;
    g1[7] = 0;
    i32x4 z4 = {0, 0, 0, 0};
#if __clang_major__ >= 23
    i32x8 z8 = {0, 0, 0, 0, 0, 0, 0, 0};
    __builtin_amdgcn_tensor_load_to_lds(g0, g1, z4, z4, z8, 0);
#else
    __builtin_amdgcn_tensor_load_to_lds(g0, g1, z4, z4, 0);
#endif
}
#endif

// ---------------------------------------------------------------------------
// C[M,N] = A[M,K] (bf16 row-major) x W[N,K]^T (bf16 row-major)
// Block = 8 waves = 128x64 tile. W panel (64x32) TDM-staged to LDS, double
// buffered; each wave computes a 16x64 strip (4 WMMA accumulators).
// A/B fragments in contiguous k order, k = 16*half + s (16-bit A 16x32 / B 32x16).
// M must be a multiple of 128 (true for all call sites) -> uniform barriers.
// ---------------------------------------------------------------------------
template <int OUT_F32>
__global__ __launch_bounds__(256) void gemm_wmma(
        const unsigned short* __restrict__ A,
        const unsigned short* __restrict__ W,
        void* __restrict__ Cout,
        int M, int N, int K, int ldc_m, int ldc_n) {
#if USE_TDM
    __shared__ __align__(16) unsigned short ldsB[2][64 * 32];
#endif
    const int lane   = threadIdx.x & 31;
    const int w      = threadIdx.x >> 5;
    const int tilesN = N >> 6;
    const int bm     = blockIdx.x / tilesN;
    const int bn     = blockIdx.x % tilesN;
    const int m0     = bm * 128 + w * 16;
    const int n0     = bn * 64;
    const int half   = lane >> 4;
    const int l16    = lane & 15;

    const unsigned short* aRow = A + (long)(m0 + l16) * K + half * 16;
    v8f acc[4] = {};

#if USE_TDM
    const unsigned short* wPanel = W + (long)n0 * K;   // 64 rows of W
    if (w == 0) tdm_load_panel((unsigned)(unsigned long long)(void*)&ldsB[0][0],
                               wPanel, K, K);
    const int nsteps = K >> 5;
    for (int ks = 0; ks < nsteps; ++ks) {
        const int k0  = ks << 5;
        const int cur = ks & 1;
        if (w == 0) __builtin_amdgcn_s_wait_tensorcnt(0);
        __syncthreads();                               // panel `cur` ready
        if (w == 0 && ks + 1 < nsteps)
            tdm_load_panel((unsigned)(unsigned long long)(void*)&ldsB[cur ^ 1][0],
                           wPanel + k0 + 32, K, K);
        FragAB a;
        a.q[0] = *(const u32x4*)(aRow + k0);
        a.q[1] = *(const u32x4*)(aRow + k0 + 8);
        __builtin_prefetch(aRow + k0 + 64, 0, 1);      // global_prefetch_b8
#pragma unroll
        for (int t = 0; t < 4; ++t) {
            FragAB b;                                  // ds_load_b128 x2
            const unsigned short* p = &ldsB[cur][(t * 16 + l16) * 32 + half * 16];
            b.q[0] = *(const u32x4*)(p);
            b.q[1] = *(const u32x4*)(p + 8);
            acc[t] = __builtin_amdgcn_wmma_f32_16x16x32_bf16(
                false, a.v, false, b.v, (short)0, acc[t], false, false);
        }
        __syncthreads();                               // panel `cur` consumed
    }
#else
    const unsigned short* wRow = W + (long)(n0 + l16) * K + half * 16;
    for (int k0 = 0; k0 < K; k0 += 32) {
        FragAB a;
        a.q[0] = *(const u32x4*)(aRow + k0);
        a.q[1] = *(const u32x4*)(aRow + k0 + 8);
        __builtin_prefetch(aRow + k0 + 64, 0, 1);
#pragma unroll
        for (int t = 0; t < 4; ++t) {
            FragAB b;
            const unsigned short* p = wRow + (long)t * 16 * K + k0;
            b.q[0] = *(const u32x4*)(p);
            b.q[1] = *(const u32x4*)(p + 8);
            acc[t] = __builtin_amdgcn_wmma_f32_16x16x32_bf16(
                false, a.v, false, b.v, (short)0, acc[t], false, false);
        }
    }
#endif

    // C/D layout: VGPR r, lanes 0-15 -> M=r, N=lane; lanes 16-31 -> M=8+r
    const int mrow = m0 + half * 8;
#pragma unroll
    for (int t = 0; t < 4; ++t) {
#pragma unroll
        for (int r = 0; r < 8; ++r) {
            long idx = (long)(mrow + r) * ldc_m + (long)(n0 + t * 16 + l16) * ldc_n;
            if (OUT_F32) ((float*)Cout)[idx] = acc[t][r];
            else         ((unsigned short*)Cout)[idx] = f2bf(acc[t][r]);
        }
    }
}

// ---------------------------------------------------------------------------
// RoPE on Q [S][H*128] and K [S][KVH*128] (bf16, in place).
// ---------------------------------------------------------------------------
__global__ __launch_bounds__(256) void rope_kernel(unsigned short* __restrict__ Q,
                                                   unsigned short* __restrict__ Kb,
                                                   const int* __restrict__ pos) {
    int idx  = blockIdx.x * 256 + threadIdx.x;   // S * (H+KVH) * 64 threads
    int d    = idx & 63;
    int rest = idx >> 6;
    int hh   = rest % (NHEAD + NKVH);
    int s    = rest / (NHEAD + NKVH);
    if (s >= S_LEN) return;

    float p   = (float)pos[s];
    float inv = __powf(10000.0f, -((float)(2 * d)) * (1.0f / 128.0f));
    float sn, cs;
    __sincosf(p * inv, &sn, &cs);

    unsigned short* buf;
    long off;
    if (hh < NHEAD) { buf = Q;  off = (long)s * HID   + hh * HDIM + d; }
    else            { buf = Kb; off = (long)s * KVHID + (hh - NHEAD) * HDIM + d; }
    float x1 = bf2f(buf[off]);
    float x2 = bf2f(buf[off + 64]);
    buf[off]      = f2bf(x1 * cs - x2 * sn);
    buf[off + 64] = f2bf(x2 * cs + x1 * sn);
}

// ---------------------------------------------------------------------------
// Flash attention: one wave per (head, 16-query tile). Causal.
// Q [S][H*128], K [S][KVH*128], Vt [KVH*128][S] (transposed), O [S][H*128].
// Per 32-key step: 8 QK WMMAs -> online softmax (ds_swizzle row reductions in
// the 16-lane C-layout groups) -> P through per-wave LDS (C->A transpose) ->
// 8 PV WMMAs into resident 16x128 f32 accumulators.
// ---------------------------------------------------------------------------
__global__ __launch_bounds__(128) void flash_attn(
        const unsigned short* __restrict__ Q,
        const unsigned short* __restrict__ Kb,
        const unsigned short* __restrict__ Vt,
        unsigned short* __restrict__ O) {
    __shared__ __align__(16) unsigned short ldsP[4][16 * 32];

    const int lane = threadIdx.x & 31;
    const int w    = threadIdx.x >> 5;
    const int tid  = blockIdx.x * 4 + w;
    const int h    = tid >> 7;              // 0..31
    const int q0   = (tid & 127) << 4;      // 0..2032
    const int kvh  = h >> 2;
    const int half = lane >> 4;
    const int l16  = lane & 15;
    const int mrow = q0 + half * 8;         // first of this lane's 8 rows

    FragAB qf[4];                            // resident Q fragments
    {
        const unsigned short* qrow = Q + (long)(q0 + l16) * HID + h * HDIM + half * 16;
#pragma unroll
        for (int t = 0; t < 4; ++t) {
            qf[t].q[0] = *(const u32x4*)(qrow + t * 32);
            qf[t].q[1] = *(const u32x4*)(qrow + t * 32 + 8);
        }
    }

    v8f o[8] = {};
    float mi[8], li[8];
#pragma unroll
    for (int r = 0; r < 8; ++r) { mi[r] = -1e30f; li[r] = 0.0f; }
    const float scale = 0.08838834764831845f;   // 1/sqrt(128)
    unsigned short* pbuf = ldsP[w];

    for (int j0 = 0; j0 < q0 + 16; j0 += 32) {
        float sc[2][8];
#pragma unroll
        for (int sub = 0; sub < 2; ++sub) {
            const int k0 = j0 + sub * 16;
            if (k0 < q0 + 16) {
                v8f s = {};
                const unsigned short* krow =
                    Kb + (long)(k0 + l16) * KVHID + kvh * HDIM + half * 16;
#pragma unroll
                for (int t = 0; t < 4; ++t) {
                    FragAB kf;
                    kf.q[0] = *(const u32x4*)(krow + t * 32);
                    kf.q[1] = *(const u32x4*)(krow + t * 32 + 8);
                    s = __builtin_amdgcn_wmma_f32_16x16x32_bf16(
                        false, qf[t].v, false, kf.v, (short)0, s, false, false);
                }
#pragma unroll
                for (int r = 0; r < 8; ++r) {
                    int m = mrow + r, n = k0 + l16;
                    sc[sub][r] = (n <= m) ? s[r] * scale : -1e30f;
                }
            } else {
#pragma unroll
                for (int r = 0; r < 8; ++r) sc[sub][r] = -1e30f;
            }
        }

        // row max across the 16 lanes of each half
        float rm[8];
#pragma unroll
        for (int r = 0; r < 8; ++r) {
            rm[r] = fmaxf(sc[0][r], sc[1][r]);
            SWZ_MAX(rm[r], 1); SWZ_MAX(rm[r], 2);
            SWZ_MAX(rm[r], 4); SWZ_MAX(rm[r], 8);
        }

        float alpha[8], ps0[8], ps1[8], rs[8];
#pragma unroll
        for (int r = 0; r < 8; ++r) {
            float mn = fmaxf(mi[r], rm[r]);
            alpha[r] = __expf(mi[r] - mn);
            mi[r]    = mn;
            ps0[r]   = __expf(sc[0][r] - mn);
            ps1[r]   = __expf(sc[1][r] - mn);
            rs[r]    = ps0[r] + ps1[r];
            SWZ_ADD(rs[r], 1); SWZ_ADD(rs[r], 2);
            SWZ_ADD(rs[r], 4); SWZ_ADD(rs[r], 8);
            li[r] = li[r] * alpha[r] + rs[r];
        }
#pragma unroll
        for (int t = 0; t < 8; ++t)
#pragma unroll
            for (int r = 0; r < 8; ++r) o[t][r] *= alpha[r];

        // P (C layout) -> LDS -> P (A-fragment layout). Same-wave: DScnt ordering.
#pragma unroll
        for (int r = 0; r < 8; ++r) {
            pbuf[(half * 8 + r) * 32 + l16]      = f2bf(ps0[r]);
            pbuf[(half * 8 + r) * 32 + 16 + l16] = f2bf(ps1[r]);
        }
        FragAB pf;
        const unsigned short* prow = pbuf + l16 * 32 + half * 16;
        pf.q[0] = *(const u32x4*)(prow);
        pf.q[1] = *(const u32x4*)(prow + 8);

        // PV: B columns are V^T rows -> contiguous along keys
#pragma unroll
        for (int t = 0; t < 8; ++t) {
            FragAB vf;
            const unsigned short* vrow =
                Vt + (long)(kvh * HDIM + t * 16 + l16) * S_LEN + j0 + half * 16;
            vf.q[0] = *(const u32x4*)(vrow);
            vf.q[1] = *(const u32x4*)(vrow + 8);
            o[t] = __builtin_amdgcn_wmma_f32_16x16x32_bf16(
                false, pf.v, false, vf.v, (short)0, o[t], false, false);
        }
    }

    float linv[8];
#pragma unroll
    for (int r = 0; r < 8; ++r) linv[r] = 1.0f / li[r];
#pragma unroll
    for (int t = 0; t < 8; ++t)
#pragma unroll
        for (int r = 0; r < 8; ++r)
            O[(long)(mrow + r) * HID + h * HDIM + t * 16 + l16] =
                f2bf(o[t][r] * linv[r]);
}

// ---------------------------------------------------------------------------
// launch
// ---------------------------------------------------------------------------
extern "C" void kernel_launch(void* const* d_in, const int* in_sizes, int n_in,
                              void* d_out, int out_size, void* d_ws, size_t ws_size,
                              hipStream_t stream) {
    const float* hidden = (const float*)d_in[0];
    // d_in[1] = attention_mask: provably causal tril -> applied analytically
    const int*   pos    = (const int*)d_in[2];
    const float* Wq     = (const float*)d_in[3];
    const float* Wk     = (const float*)d_in[4];
    const float* Wv     = (const float*)d_in[5];
    const float* Wo     = (const float*)d_in[6];
    float*       out    = (float*)d_out;

    // bf16 workspace layout (element offsets)
    unsigned short* hb  = (unsigned short*)d_ws;                 //  8M  hidden
    unsigned short* wqb = hb  + (long)S_LEN * HID;               // 16M  Wq
    unsigned short* wkb = wqb + (long)HID * HID;                 //  4M  Wk
    unsigned short* wvb = wkb + (long)KVHID * HID;               //  4M  Wv
    unsigned short* wob = wvb + (long)KVHID * HID;               // 16M  Wo
    unsigned short* qb  = wob + (long)HID * HID;                 //  8M  Q
    unsigned short* kb  = qb  + (long)S_LEN * HID;               //  2M  K
    unsigned short* vtb = kb  + (long)S_LEN * KVHID;             //  2M  V^T
    unsigned short* ab  = vtb + (long)S_LEN * KVHID;             //  8M  attn out

    auto cvt = [&](const float* src, unsigned short* dst, long n) {
        cvt_bf16<<<(int)(n / 1024), 256, 0, stream>>>(src, dst, (int)n);
    };
    cvt(hidden, hb,  (long)S_LEN * HID);
    cvt(Wq,     wqb, (long)HID * HID);
    cvt(Wk,     wkb, (long)KVHID * HID);
    cvt(Wv,     wvb, (long)KVHID * HID);
    cvt(Wo,     wob, (long)HID * HID);

    // Q = hb x Wq^T   -> [S][4096] bf16   (16 x 64 blocks of 128x64)
    gemm_wmma<0><<<(S_LEN / 128) * (HID / 64), 256, 0, stream>>>(
        hb, wqb, qb, S_LEN, HID, HID, HID, 1);
    // K = hb x Wk^T   -> [S][1024] bf16
    gemm_wmma<0><<<(S_LEN / 128) * (KVHID / 64), 256, 0, stream>>>(
        hb, wkb, kb, S_LEN, KVHID, HID, KVHID, 1);
    // V^T = (hb x Wv^T)^T -> [1024][S] bf16 (transposed store)
    gemm_wmma<0><<<(S_LEN / 128) * (KVHID / 64), 256, 0, stream>>>(
        hb, wvb, vtb, S_LEN, KVHID, HID, 1, S_LEN);

    // RoPE on Q and K
    {
        long threads = (long)S_LEN * (NHEAD + NKVH) * 64;
        rope_kernel<<<(int)(threads / 256), 256, 0, stream>>>(qb, kb, pos);
    }

    // flash attention: 32 heads x 128 query tiles, 4 waves/block
    flash_attn<<<(NHEAD * (S_LEN / 16)) / 4, 128, 0, stream>>>(qb, kb, vtb, ab);

    // out = ab x Wo^T -> fp32
    gemm_wmma<1><<<(S_LEN / 128) * (HID / 64), 256, 0, stream>>>(
        ab, wob, out, S_LEN, HID, HID, HID, 1);
}
